// MultiLatentSpaceSimilarity_34986803593696
// MI455X (gfx1250) — matrix operand order repaired and checked by reading, not compiled, not run on hardware
//
#include <hip/hip_runtime.h>
#include <hip/hip_bf16.h>

typedef __attribute__((ext_vector_type(16))) __bf16 v16bf;
typedef __attribute__((ext_vector_type(8)))  __bf16 v8bf;
typedef __attribute__((ext_vector_type(8)))  float  v8f;

// Problem constants (from reference)
constexpr int B_  = 8;     // batch
constexpr int S_  = 1024;  // sequence
constexpr int C_  = 128;   // channels
constexpr int G_  = 3;     // groups
constexpr int H_  = 4;     // heads
constexpr int D_  = 32;    // dim_head  (== WMMA K depth!)
constexpr int P_  = 8;     // prototypes per group (SPLITS)
constexpr int L_  = 64;    // latent length
constexpr int NI_ = 128;   // H_*D_
__device__ constexpr float SCALE_ = 0.17677669529663687f; // 32^-0.5

// ---------------------------------------------------------------------------
// init: zero heat + dist accumulators (must run every call; atomics accumulate)
// ---------------------------------------------------------------------------
__global__ void init_kernel(float* __restrict__ heat, float* __restrict__ dist) {
    int i = blockIdx.x * 256 + threadIdx.x;
    if (i < G_ * B_ * P_ * S_) heat[i] = 0.0f;
    if (i < B_ * G_ * P_)      dist[i] = 0.0f;
}

// ---------------------------------------------------------------------------
// proj_kv: K/V projections.  One wave per 16x16 (i,s) tile; A = weights (M=i),
// B = x (N=s), K = c (128 -> 4 bf16 WMMA chunks).  x fragment shared by K & V.
// kbuf layout [g][b][h][s][d] (d contiguous), vT layout [g][b][h][d][s].
// ---------------------------------------------------------------------------
__global__ void __launch_bounds__(256)
proj_kv_kernel(const float* __restrict__ x,
               const float* __restrict__ kw, const float* __restrict__ vw,
               __bf16* __restrict__ kbuf, __bf16* __restrict__ vT) {
    const int tid = threadIdx.x;
    const int wave = tid >> 5, lane = tid & 31;
    const int m = lane & 15, half = lane >> 4;

    const int tile = blockIdx.x * 8 + wave;        // 12288 tiles total
    const int g  = tile / (B_ * 512);
    const int rem = tile % (B_ * 512);
    const int b  = rem >> 9;
    const int t2 = rem & 511;
    const int st = t2 >> 3;                        // s tile (64)
    const int it = t2 & 7;                         // i tile (8)

    const float* kwr = kw + (size_t)g * C_ * C_ + (it * 16 + m) * C_;
    const float* vwr = vw + (size_t)g * C_ * C_ + (it * 16 + m) * C_;
    const int s = st * 16 + m;
    const float* xcol = x + (size_t)b * C_ * S_ + s;

    v8f acck = {};
    v8f accv = {};
#pragma unroll
    for (int c0 = 0; c0 < C_; c0 += 32) {
        v16bf ak, av, bx;
#pragma unroll
        for (int e = 0; e < 8; ++e) {
            ak[e]     = (__bf16)kwr[c0 + half * 8 + e];
            ak[e + 8] = (__bf16)kwr[c0 + 16 + half * 8 + e];
            av[e]     = (__bf16)vwr[c0 + half * 8 + e];
            av[e + 8] = (__bf16)vwr[c0 + 16 + half * 8 + e];
        }
#pragma unroll
        for (int e = 0; e < 16; ++e)
            bx[e] = (__bf16)xcol[(size_t)(c0 + ((e & 8) << 1) + half * 8 + (e & 7)) * S_];
        acck = __builtin_amdgcn_wmma_f32_16x16x32_bf16(false, ak, false, bx,
                                                       (short)0, acck, false, false);
        accv = __builtin_amdgcn_wmma_f32_16x16x32_bf16(false, av, false, bx,
                                                       (short)0, accv, false, false);
    }

    // D layout: lane (n=m, half), VGPR r -> i = it*16 + 8*half + r, col s.
    const int i0 = it * 16 + half * 8;
    const int h  = i0 >> 5, d0 = i0 & 31;
    const size_t base = (size_t)((g * B_ + b) * H_ + h);

    v8bf kv;
#pragma unroll
    for (int r = 0; r < 8; ++r) kv[r] = (__bf16)acck[r];
    *(v8bf*)(kbuf + (base * S_ + s) * D_ + d0) = kv;

    __bf16* vdst = vT + (base * D_ + d0) * S_ + s;
#pragma unroll
    for (int r = 0; r < 8; ++r) vdst[(size_t)r * S_] = (__bf16)accv[r];
}

// ---------------------------------------------------------------------------
// proj_q: q_proj = prototypes^T x q_w^T.  One wave per 16x16 (i, l) tile.
// Emits fp32 q_proj [g][p][l][i] and bf16 qbuf [g][h][p][l][d].
// ---------------------------------------------------------------------------
__global__ void __launch_bounds__(256)
proj_q_kernel(const float* __restrict__ prot, const float* __restrict__ qw,
              __bf16* __restrict__ qbuf, float* __restrict__ qproj) {
    const int tid = threadIdx.x;
    const int wave = tid >> 5, lane = tid & 31;
    const int m = lane & 15, half = lane >> 4;

    const int tile = blockIdx.x * 8 + wave;        // 768 tiles
    const int g  = tile >> 8;
    const int r0 = tile & 255;
    const int p  = r0 >> 5;
    const int r1 = r0 & 31;
    const int lt = r1 >> 3;                        // l tile (4)
    const int it = r1 & 7;                         // i tile (8)

    const float* qwr  = qw + (size_t)g * C_ * C_ + (it * 16 + m) * C_;
    const float* pcol = prot + (size_t)(g * P_ + p) * C_ * L_ + (lt * 16 + m);

    v8f acc = {};
#pragma unroll
    for (int c0 = 0; c0 < C_; c0 += 32) {
        v16bf a, bp;
#pragma unroll
        for (int e = 0; e < 8; ++e) {
            a[e]     = (__bf16)qwr[c0 + half * 8 + e];
            a[e + 8] = (__bf16)qwr[c0 + 16 + half * 8 + e];
        }
#pragma unroll
        for (int e = 0; e < 16; ++e)
            bp[e] = (__bf16)pcol[(size_t)(c0 + ((e & 8) << 1) + half * 8 + (e & 7)) * L_];
        acc = __builtin_amdgcn_wmma_f32_16x16x32_bf16(false, a, false, bp,
                                                      (short)0, acc, false, false);
    }

    const int i0 = it * 16 + half * 8;
    const int h  = i0 >> 5, d0 = i0 & 31;
    const int l  = lt * 16 + m;

    float* qpdst = qproj + ((size_t)(g * P_ + p) * L_ + l) * NI_ + i0;
#pragma unroll
    for (int r = 0; r < 8; ++r) qpdst[r] = acc[r];

    v8bf qv;
#pragma unroll
    for (int r = 0; r < 8; ++r) qv[r] = (__bf16)acc[r];
    *(v8bf*)(qbuf + (((size_t)(g * H_ + h) * P_ + p) * L_ + l) * D_ + d0) = qv;
}

// ---------------------------------------------------------------------------
// attn: one workgroup (8 waves) per (g,b,h,p).  Full 64x1024 score matrix in
// LDS (256KB of the WGP's 320KB).  dots -> softmax -> heat -> attn@V + dist.
// All WMMA-reaching control flow is wave-uniform (EXEC all ones).
// ---------------------------------------------------------------------------
__global__ void __launch_bounds__(256)
attn_kernel(const __bf16* __restrict__ kbuf, const __bf16* __restrict__ vT,
            const __bf16* __restrict__ qbuf, const float* __restrict__ qproj,
            float* __restrict__ heat, float* __restrict__ dist_acc) {
    __shared__ float sm[64][1024];     // 256 KB: scores, then attn weights
    __shared__ float redA[64][4];
    __shared__ float redB[64][4];

    const int tid = threadIdx.x;
    const int wave = tid >> 5, lane = tid & 31;
    const int m = lane & 15, half = lane >> 4;

    const int bid = blockIdx.x;                    // 768 blocks
    const int g  = bid >> 8;
    const int r0 = bid & 255;
    const int b  = r0 >> 5;
    const int r1 = r0 & 31;
    const int h  = r1 >> 3, p = r1 & 7;

    const __bf16* kb = kbuf + (size_t)((g * B_ + b) * H_ + h) * S_ * D_;
    const __bf16* vb = vT   + (size_t)((g * B_ + b) * H_ + h) * D_ * S_;
    const __bf16* qb = qbuf + (size_t)((g * H_ + h) * P_ + p) * L_ * D_;

    // Q A-fragments for the 4 l-tiles (K = D = 32, one WMMA per score tile)
    v16bf aq[4];
#pragma unroll
    for (int lt = 0; lt < 4; ++lt) {
        const __bf16* qrow = qb + (lt * 16 + m) * D_;
#pragma unroll
        for (int e = 0; e < 8; ++e) {
            aq[lt][e]     = qrow[half * 16 + e];
            aq[lt][e + 8] = qrow[half * 16 + 8 + e];
        }
    }

    // dots = (q . K^T) * SCALE  ->  LDS
    for (int st = wave; st < 64; st += 8) {        // uniform: 8 iters per wave
        const int s = st * 16 + m;
        v16bf bk = *(const v16bf*)(kb + (size_t)s * D_ + half * 16);
#pragma unroll
        for (int lt = 0; lt < 4; ++lt) {
            v8f acc = {};
            acc = __builtin_amdgcn_wmma_f32_16x16x32_bf16(false, aq[lt], false, bk,
                                                          (short)0, acc, false, false);
#pragma unroll
            for (int r = 0; r < 8; ++r)
                sm[lt * 16 + half * 8 + r][s] = acc[r] * SCALE_;
        }
    }
    __syncthreads();

    // softmax over S per row l: 4 threads per row
    {
        const int l = tid >> 2, qd = tid & 3, s0 = qd << 8;
        float mx = -3.0e38f;
        for (int s2 = s0; s2 < s0 + 256; ++s2) mx = fmaxf(mx, sm[l][s2]);
        redA[l][qd] = mx;
        __syncthreads();
        mx = fmaxf(fmaxf(redA[l][0], redA[l][1]), fmaxf(redA[l][2], redA[l][3]));
        float sum = 0.0f;
        for (int s2 = s0; s2 < s0 + 256; ++s2) {
            float e2 = __expf(sm[l][s2] - mx);
            sm[l][s2] = e2;
            sum += e2;
        }
        redB[l][qd] = sum;
        __syncthreads();
        const float inv = 1.0f / (redB[l][0] + redB[l][1] + redB[l][2] + redB[l][3]);
        for (int s2 = s0; s2 < s0 + 256; ++s2) sm[l][s2] *= inv;
    }
    __syncthreads();

    // heat[g][b][p][s] += sum_l attn[l][s]   (1/(H*L) scale is argmax-invariant)
    {
        float* hp = heat + (size_t)((g * B_ + b) * P_ + p) * S_;
        for (int sc = tid; sc < S_; sc += 256) {   // uniform: 4 iters each
            float hs = 0.0f;
            for (int l2 = 0; l2 < 64; ++l2) hs += sm[l2][sc];
            atomicAdd(hp + sc, hs);
        }
    }

    // out = attn @ V (per-wave 16x16 tile, K=1024 -> 32 WMMA chunks), fused
    // partial sum of (q_proj - out)^2 into dist accumulator.
    {
        const int lt = wave >> 1, dt = wave & 1;
        const int l    = lt * 16 + m;              // A row for this lane
        const int dloc = dt * 16 + m;              // B col for this lane
        v8f acc = {};
        for (int kc = 0; kc < 32; ++kc) {
            const int sb = kc * 32;
            v16bf aa;
#pragma unroll
            for (int e = 0; e < 8; ++e) {
                aa[e]     = (__bf16)sm[l][sb + half * 16 + e];
                aa[e + 8] = (__bf16)sm[l][sb + half * 16 + 8 + e];
            }
            v16bf bv = *(const v16bf*)(vb + (size_t)dloc * S_ + sb + half * 16);
            acc = __builtin_amdgcn_wmma_f32_16x16x32_bf16(false, aa, false, bv,
                                                          (short)0, acc, false, false);
        }
        const int i = h * 32 + dloc;
        float local = 0.0f;
#pragma unroll
        for (int r = 0; r < 8; ++r) {
            const int lrow = lt * 16 + half * 8 + r;
            float diff = qproj[((size_t)(g * P_ + p) * L_ + lrow) * NI_ + i] - acc[r];
            local += diff * diff;
        }
        for (int off = 16; off; off >>= 1) local += __shfl_down(local, off);
        if (lane == 0) atomicAdd(dist_acc + b * (G_ * P_) + g * P_ + p, local);
    }
}

// ---------------------------------------------------------------------------
// finalize: dist /= (L*NI); argmax over S of heat -> index (written as float).
// One wave per (b, j=g*8+p).
// ---------------------------------------------------------------------------
__global__ void finalize_kernel(const float* __restrict__ heat,
                                const float* __restrict__ dist_acc,
                                float* __restrict__ out) {
    const int blk = blockIdx.x;                    // 192
    const int b = blk / (G_ * P_), j = blk % (G_ * P_);
    const int g = j >> 3, p = j & 7;
    const float* hrow = heat + (size_t)((g * B_ + b) * P_ + p) * S_;
    const int lane = threadIdx.x;

    float best = -3.0e38f;
    int bi = 0;
    for (int s = lane; s < S_; s += 32) {
        float v = hrow[s];
        if (v > best) { best = v; bi = s; }
    }
    for (int off = 16; off; off >>= 1) {
        float ov = __shfl_down(best, off);
        int   oi = __shfl_down(bi, off);
        if (ov > best || (ov == best && oi < bi)) { best = ov; bi = oi; }
    }
    if (lane == 0) {
        out[b * (G_ * P_) + j] = dist_acc[b * (G_ * P_) + j] * (1.0f / 8192.0f);
        out[B_ * (G_ * P_) + b * (G_ * P_) + j] = (float)bi;
    }
}

// ---------------------------------------------------------------------------
extern "C" void kernel_launch(void* const* d_in, const int* in_sizes, int n_in,
                              void* d_out, int out_size, void* d_ws, size_t ws_size,
                              hipStream_t stream) {
    const float* x    = (const float*)d_in[0];  // (8,128,1024)
    const float* prot = (const float*)d_in[1];  // (24,128,64)
    const float* qw   = (const float*)d_in[2];  // (3,128,128)
    const float* kw   = (const float*)d_in[3];  // (3,128,128)
    const float* vw   = (const float*)d_in[4];  // (3,128,128)
    float* out = (float*)d_out;                 // 192 dist + 192 idx

    // Workspace layout (~14.6 MB total)
    constexpr size_t KV_ELEMS   = (size_t)G_ * B_ * H_ * S_ * D_;  // 3,145,728
    constexpr size_t QB_ELEMS   = (size_t)G_ * H_ * P_ * L_ * D_;  //   196,608
    constexpr size_t QP_ELEMS   = (size_t)G_ * P_ * L_ * NI_;      //   196,608
    constexpr size_t HEAT_ELEMS = (size_t)G_ * B_ * P_ * S_;       //   196,608

    char* ws = (char*)d_ws;
    __bf16* kbuf = (__bf16*)ws;
    __bf16* vT   = kbuf + KV_ELEMS;
    __bf16* qbuf = vT + KV_ELEMS;
    float*  qproj = (float*)(ws + (2 * KV_ELEMS + QB_ELEMS) * sizeof(__bf16));
    float*  heat  = qproj + QP_ELEMS;
    float*  dist  = heat + HEAT_ELEMS;

    init_kernel<<<(HEAT_ELEMS + 255) / 256, 256, 0, stream>>>(heat, dist);
    proj_kv_kernel<<<(G_ * B_ * 512) / 8, 256, 0, stream>>>(x, kw, vw, kbuf, vT);
    proj_q_kernel<<<(G_ * P_ * 4 * 8) / 8, 256, 0, stream>>>(prot, qw, qbuf, qproj);
    attn_kernel<<<G_ * B_ * H_ * P_, 256, 0, stream>>>(kbuf, vT, qbuf, qproj, heat, dist);
    finalize_kernel<<<B_ * G_ * P_, 32, 0, stream>>>(heat, dist, out);
}
